// GCN_10479720202240
// MI455X (gfx1250) — matrix operand order repaired and checked by reading, not compiled, not run on hardware
//
#include <hip/hip_runtime.h>
#include <hip/hip_bf16.h>

// ---------------------------------------------------------------------------
// 2-layer GCN for MI455X (gfx1250, wave32).
//  N=100000 nodes, E=3.2M edges, F_in=256, H=16, F_out=2.
//  GEMM1 (x @ W1) runs on v_wmma_f32_16x16x32_f16, one 16x16 tile per wave.
//  W1 (16KB) is staged once per block into LDS, pre-converted to f16 and
//  pre-swizzled into B-operand lane layout, so each wave reads each B
//  fragment as one aligned 32B LDS access (2x ds_load_b128) instead of
//  16 scalar global loads. Edge aggregation uses L2-resident
//  global_atomic_add_f32 (agg buffers are 6.4MB << 192MB L2).
// ---------------------------------------------------------------------------

typedef __attribute__((ext_vector_type(16))) _Float16 v16h;
typedef __attribute__((ext_vector_type(8)))  float    v8f;
typedef __attribute__((ext_vector_type(4)))  float    v4f;

#define IN_FEATS 256
#define HIDDEN   16
#define OUT_FEATS 2

// --- init: deg=1.0 (self loop), agg1=0, out=0 ------------------------------
__global__ void gcn_init(float* __restrict__ deg, float* __restrict__ agg1,
                         float* __restrict__ out, int N) {
    long long tid = (long long)blockIdx.x * blockDim.x + threadIdx.x;
    long long n16 = (long long)N * HIDDEN;
    if (tid < n16) agg1[tid] = 0.0f;
    if (tid < (long long)N * OUT_FEATS) out[tid] = 0.0f;
    if (tid < N) deg[tid] = 1.0f;
}

// --- degree accumulate over edge destinations ------------------------------
__global__ void gcn_degree(const int* __restrict__ dst, float* __restrict__ deg,
                           int E) {
    long long e = (long long)blockIdx.x * blockDim.x + threadIdx.x;
    if (e >= E) return;
    atomicAdd(&deg[dst[e]], 1.0f);
}

// --- deg -> dinv = rsqrt(deg) in place (deg >= 1 always) -------------------
__global__ void gcn_rsqrt(float* __restrict__ deg, int N) {
    long long i = (long long)blockIdx.x * blockDim.x + threadIdx.x;
    if (i >= N) return;
    deg[i] = rsqrtf(deg[i]);
}

// --- GEMM1: h[N,16] = x[N,256] @ W1[256,16] via WMMA f16 -------------------
// One wave computes one 16(M)x16(N) tile; K=256 -> 8 wmma steps of K=32.
// A-layout (16-bit A 16x32, same K order as B):
//   lane L: M = L&15, kgrp = L>>4; elem j -> K = kgrp*16 + j
// B-layout (K x 16): lane L: Ncol = L&15; elem j -> K = kgrp*16 + j
// C/D-layout: elem r -> M = r + 8*kgrp, Ncol = L&15.
//
// LDS staging layout for B (f16): Blds[((kk*2 + kgrp)*16 + col)*16 + j]
//   = W1[(kk*32 + kgrp*16 + j)*16 + col]
// -> each lane's per-step fragment is 16 contiguous halves (32B, aligned).
__global__ void __launch_bounds__(256)
gcn_gemm1_wmma(const float* __restrict__ x,
               const float* __restrict__ W1,
               float* __restrict__ h, int nTiles) {
    __shared__ _Float16 Blds[IN_FEATS * HIDDEN];   // 8 KB

    // ---- cooperative fill: thread t repacks W1 row k = t (16 floats) ----
    {
        const int k    = threadIdx.x;              // 0..255 == K index
        const int kk   = k >> 5;
        const int kgrp = (k >> 4) & 1;
        const int j    = k & 15;
        const v4f* wrow = (const v4f*)(W1 + (size_t)k * HIDDEN);
        v4f w0 = wrow[0], w1 = wrow[1], w2 = wrow[2], w3 = wrow[3];
        _Float16* base = &Blds[(size_t)((kk * 2 + kgrp) * 16) * 16 + j];
#pragma unroll
        for (int n = 0; n < 4; ++n) {
            base[(n + 0)  * 16] = (_Float16)w0[n];
            base[(n + 4)  * 16] = (_Float16)w1[n];
            base[(n + 8)  * 16] = (_Float16)w2[n];
            base[(n + 12) * 16] = (_Float16)w3[n];
        }
    }
    __syncthreads();

    const int waveInBlk = threadIdx.x >> 5;
    const int lane      = threadIdx.x & 31;
    const int tile      = blockIdx.x * (blockDim.x >> 5) + waveInBlk; // wave-uniform
    if (tile >= nTiles) return;   // uniform branch: EXEC stays all-ones

    const int row0 = tile * 16;
    const int col  = lane & 15;
    const int kgrp = lane >> 4;           // 0 or 1
    const int arow = row0 + col;          // A: M = lane&15

    v8f acc = {};
#pragma unroll
    for (int kk = 0; kk < 8; ++kk) {
        const int k0 = kk * 32;
        // ---- A operand: one contiguous 16-float run -> 4x global b128 ----
        const v4f* ap = (const v4f*)(x + (size_t)arow * IN_FEATS + k0 + kgrp * 16);
        v4f a0 = ap[0];           // K = k0 + kgrp*16 + 0..3
        v4f a1 = ap[1];           //                  + 4..7
        v4f a2 = ap[2];           // K = k0 + kgrp*16 + 8..11
        v4f a3 = ap[3];           //                  + 12..15
        v16h a;
#pragma unroll
        for (int j = 0; j < 4; ++j) {
            a[j]      = (_Float16)a0[j];
            a[4 + j]  = (_Float16)a1[j];
            a[8 + j]  = (_Float16)a2[j];
            a[12 + j] = (_Float16)a3[j];
        }
        // ---- B operand: one aligned 32B LDS read (2x ds_load_b128) ----
        v16h b = *(const v16h*)&Blds[(size_t)((kk * 2 + kgrp) * 16 + col) * 16];

        acc = __builtin_amdgcn_wmma_f32_16x16x32_f16(
            /*neg_a=*/false, a, /*neg_b=*/false, b,
            /*c_mod=*/(short)0, acc, /*reuse_a=*/false, /*reuse_b=*/false);
    }

    // ---- store D: elem r -> row row0 + 8*kgrp + r, column col ----
    float* hp = h + (size_t)(row0 + kgrp * 8) * HIDDEN + col;
#pragma unroll
    for (int r = 0; r < 8; ++r) hp[r * HIDDEN] = acc[r];
}

// --- edge scatter, 16 feats/edge: agg[dst] += h[src]*dinv[s]*dinv[d] -------
__global__ void gcn_scatter16(const int* __restrict__ src,
                              const int* __restrict__ dst,
                              const float* __restrict__ dinv,
                              const float* __restrict__ h,
                              float* __restrict__ agg, long long nWork) {
    long long tid = (long long)blockIdx.x * blockDim.x + threadIdx.x;
    if (tid >= nWork) return;
    int f = (int)(tid & (HIDDEN - 1));
    long long e = tid >> 4;
    int s = src[e], d = dst[e];
    float w = dinv[s] * dinv[d];
    atomicAdd(&agg[(size_t)d * HIDDEN + f], h[(size_t)s * HIDDEN + f] * w);
}

// --- epilogue 1: h1 = relu(agg1 + h*selfnorm + b1), in place on agg1 -------
__global__ void gcn_post1(float* __restrict__ agg1, const float* __restrict__ h,
                          const float* __restrict__ dinv,
                          const float* __restrict__ b1, int N) {
    long long tid = (long long)blockIdx.x * blockDim.x + threadIdx.x;
    if (tid >= (long long)N * HIDDEN) return;
    int f = (int)(tid & (HIDDEN - 1));
    long long v = tid >> 4;
    float sn = dinv[v] * dinv[v];
    float val = agg1[tid] + h[tid] * sn + b1[f];
    agg1[tid] = val > 0.0f ? val : 0.0f;
}

// --- GEMM2: z[N,2] = h1[N,16] @ W2[16,2] (6.4 MFLOP -> plain VALU) ---------
__global__ void gcn_gemm2(const float* __restrict__ h1,
                          const float* __restrict__ W2,
                          float* __restrict__ z, int N) {
    long long v = (long long)blockIdx.x * blockDim.x + threadIdx.x;
    if (v >= N) return;
    const float* r = h1 + (size_t)v * HIDDEN;
    float a0 = 0.0f, a1 = 0.0f;
#pragma unroll
    for (int k = 0; k < HIDDEN; ++k) {
        float t = r[k];
        a0 += t * W2[k * OUT_FEATS + 0];
        a1 += t * W2[k * OUT_FEATS + 1];
    }
    z[v * OUT_FEATS + 0] = a0;
    z[v * OUT_FEATS + 1] = a1;
}

// --- edge scatter, 2 feats/edge into d_out ---------------------------------
__global__ void gcn_scatter2(const int* __restrict__ src,
                             const int* __restrict__ dst,
                             const float* __restrict__ dinv,
                             const float* __restrict__ z,
                             float* __restrict__ out, long long nWork) {
    long long tid = (long long)blockIdx.x * blockDim.x + threadIdx.x;
    if (tid >= nWork) return;
    int c = (int)(tid & 1);
    long long e = tid >> 1;
    int s = src[e], d = dst[e];
    float w = dinv[s] * dinv[d];
    atomicAdd(&out[(size_t)d * OUT_FEATS + c], z[(size_t)s * OUT_FEATS + c] * w);
}

// --- epilogue 2: out += z*selfnorm + b2 ------------------------------------
__global__ void gcn_post2(float* __restrict__ out, const float* __restrict__ z,
                          const float* __restrict__ dinv,
                          const float* __restrict__ b2, int N) {
    long long tid = (long long)blockIdx.x * blockDim.x + threadIdx.x;
    if (tid >= (long long)N * OUT_FEATS) return;
    int c = (int)(tid & 1);
    long long v = tid >> 1;
    out[tid] += z[tid] * dinv[v] * dinv[v] + b2[c];
}

static inline unsigned int nblk(long long work, int t) {
    return (unsigned int)((work + t - 1) / t);
}

extern "C" void kernel_launch(void* const* d_in, const int* in_sizes, int n_in,
                              void* d_out, int out_size, void* d_ws, size_t ws_size,
                              hipStream_t stream) {
    const float* x  = (const float*)d_in[0];
    const int*   ei = (const int*)d_in[1];     // harness: integer -> int32
    const float* W1 = (const float*)d_in[2];
    const float* b1 = (const float*)d_in[3];
    const float* W2 = (const float*)d_in[4];
    const float* b2 = (const float*)d_in[5];
    float* out = (float*)d_out;

    const int N = in_sizes[0] / IN_FEATS;      // 100000
    const int E = in_sizes[1] / 2;             // 3200000
    const int* src = ei;                       // edge_index[0,:]
    const int* dst = ei + E;                   // edge_index[1,:]

    // workspace layout (floats): dinv[N] | h[N*16] | agg1/h1[N*16] | z[N*2]
    float* dinv = (float*)d_ws;
    float* h    = dinv + N;
    float* agg1 = h + (size_t)N * HIDDEN;
    float* z    = agg1 + (size_t)N * HIDDEN;

    const int T = 256;

    gcn_init<<<nblk((long long)N * HIDDEN, T), T, 0, stream>>>(dinv, agg1, out, N);
    gcn_degree<<<nblk(E, T), T, 0, stream>>>(dst, dinv, E);
    gcn_rsqrt<<<nblk(N, T), T, 0, stream>>>(dinv, N);

    const int nTiles = (N + 15) / 16;          // 6250 (N % 16 == 0)
    gcn_gemm1_wmma<<<nblk((long long)nTiles * 32, T), T, 0, stream>>>(x, W1, h, nTiles);

    gcn_scatter16<<<nblk((long long)E * HIDDEN, T), T, 0, stream>>>(
        src, dst, dinv, h, agg1, (long long)E * HIDDEN);
    gcn_post1<<<nblk((long long)N * HIDDEN, T), T, 0, stream>>>(agg1, h, dinv, b1, N);

    gcn_gemm2<<<nblk(N, T), T, 0, stream>>>(agg1, W2, z, N);
    gcn_scatter2<<<nblk((long long)E * OUT_FEATS, T), T, 0, stream>>>(
        src, dst, dinv, z, out, (long long)E * OUT_FEATS);
    gcn_post2<<<nblk((long long)N * OUT_FEATS, T), T, 0, stream>>>(out, z, dinv, b2, N);
}
